// RNN_17918603559598
// MI455X (gfx1250) — matrix operand order repaired and checked by reading, not compiled, not run on hardware
//
#include <hip/hip_runtime.h>
#include <hip/hip_bf16.h>

typedef __attribute__((ext_vector_type(16))) _Float16 v16h;
typedef __attribute__((ext_vector_type(8)))  float    v8f;

#define H_DIM     300
#define HP        320            // H padded to 10 K-chunks of 32
#define KC        10             // K chunks per matmul
#define NT        19             // N tiles of 16 (covers 304 >= 300)
#define T_STEPS   2000
#define B_TOT     32
#define MT        16             // batch rows per workgroup (WMMA M)
#define NWAVES    NT
#define NTHREADS  (NWAVES * 32)

#define TAU        0.01f
#define TAU_DEPR   0.2f
#define TAU_FACIL  1.5f
#define BETA       80.0f
#define GAIN       1.0f
#define ACT_GAIN   8.0f
#define ACT_THRESH 0.5f
#define DT_C       0.001f

__device__ __forceinline__ float sigm(float x) {
    return 1.0f / (1.0f + __expf(-x));
}

__global__ __launch_bounds__(NTHREADS)
void tm_rnn_persistent(const float* __restrict__ I,
                       const float* __restrict__ h0,
                       const float* __restrict__ r0,
                       const float* __restrict__ u0,
                       const float* __restrict__ p_rel,
                       const float* __restrict__ offset_ih,
                       const float* __restrict__ W_hh,
                       const float* __restrict__ W_hz,
                       float* __restrict__ out)
{
    extern __shared__ char smem[];
    _Float16* wpack = (_Float16*)smem;                 // KC*NT*512 halves (B-frag layout)
    _Float16* preA  = wpack + KC * NT * 512;           // MT*HP halves (A row-major)
    float*    zbuf  = (float*)(preA + MT * HP);        // MT floats

    const int tid  = threadIdx.x;
    const int lane = tid & 31;
    const int wv   = tid >> 5;             // wave index == N tile index
    const int bg0  = blockIdx.x * MT;      // batch base of this workgroup

    // ---- one-time: pack W_hh^T into the WMMA B-fragment layout (f16 in LDS) ----
    // B[k][n] = W_hh[n][k]; lane l of tile nt holds column n = nt*16 + (l&15),
    // K range (l>>4)*16 .. +15 within each 32-wide chunk, contiguous halves.
    for (int idx = tid; idx < KC * NT * 512; idx += NTHREADS) {
        int i   = idx & 15;
        int lp  = (idx >> 4) & 31;
        int nt  = (idx >> 9) % NT;
        int kc  = idx / (512 * NT);
        int n_g = nt * 16 + (lp & 15);
        int k_g = kc * 32 + (lp >> 4) * 16 + i;
        float v = (n_g < H_DIM && k_g < H_DIM) ? W_hh[n_g * H_DIM + k_g] : 0.0f;
        wpack[idx] = (_Float16)v;
    }
    // zero the K padding region of the activation buffer (never rewritten)
    for (int idx = tid; idx < MT * (HP - H_DIM); idx += NTHREADS) {
        int m = idx / (HP - H_DIM);
        int k = H_DIM + idx % (HP - H_DIM);
        preA[m * HP + k] = (_Float16)0.0f;
    }
    if (tid < MT) zbuf[tid] = 0.0f;

    // ---- persistent per-lane state (WMMA C/D accumulator layout) ----
    const int col   = lane & 15;
    const int hidx  = wv * 16 + col;            // hidden unit owned by this lane
    const bool hv   = (hidx < H_DIM);
    const int mbase = (lane >> 4) * 8;          // this lane owns rows mbase..mbase+7

    float pr   = hv ? p_rel[hidx]     : 1.0f;
    float scl  = hv ? (GAIN / pr)     : 0.0f;
    float offv = hv ? offset_ih[hidx] : 0.0f;
    float whz  = hv ? W_hz[hidx]      : 0.0f;

    float rs[8], us[8], hs[8], hts[8];
#pragma unroll
    for (int q = 0; q < 8; ++q) {
        int b = bg0 + mbase + q;
        float h_i = hv ? h0[b * H_DIM + hidx] : 0.0f;
        rs[q]  = hv ? r0[b * H_DIM + hidx] : 0.0f;
        us[q]  = hv ? u0[b * H_DIM + hidx] : 0.0f;
        hs[q]  = h_i;
        hts[q] = sigm(ACT_GAIN * (h_i - ACT_THRESH));
        float pre0 = hts[q] * rs[q] * us[q] * scl;
        preA[(mbase + q) * HP + hidx] = (_Float16)(hv ? pre0 : 0.0f);
    }
    __syncthreads();

    // constant per-lane fragment addresses
    const int arow  = lane & 15;
    const int akoff = (lane >> 4) * 16;
    const _Float16* bcol = wpack + ((size_t)wv * 512 + (size_t)lane * 16);

    for (int t = 0; t < T_STEPS; ++t) {
        // ---- matmul: acc[m][hidx] = sum_k pre[m][k] * W_hh[hidx][k] ----
        v8f acc = {};
#pragma unroll
        for (int kc = 0; kc < KC; ++kc) {
            union { uint4 q[2]; v16h v; } Af, Bf;
            const _Float16* ap = preA + arow * HP + kc * 32 + akoff;
            Af.q[0] = *(const uint4*)(ap);
            Af.q[1] = *(const uint4*)(ap + 8);
            const _Float16* bp = bcol + (size_t)kc * (NT * 512);
            Bf.q[0] = *(const uint4*)(bp);
            Bf.q[1] = *(const uint4*)(bp + 8);
            acc = __builtin_amdgcn_wmma_f32_16x16x32_f16(false, Af.v, false, Bf.v,
                                                         (short)0, acc, false, false);
        }
        __syncthreads();   // all A-fragment reads done; prev z write/reset done

        // ---- Euler update of Tsodyks-Markram state, write outputs ----
        float zpart[8];
#pragma unroll
        for (int q = 0; q < 8; ++q) {
            int m = mbase + q;
            int b = bg0 + m;
            float It = hv ? I[((size_t)b * T_STEPS + t) * H_DIM + hidx] : 0.0f;
            if (hv && t + 1 < T_STEPS)
                __builtin_prefetch(&I[((size_t)b * T_STEPS + t + 1) * H_DIM + hidx], 0, 0);
            float r = rs[q], u = us[q], h = hs[q], ht = hts[q];
            float r_new = r + ((1.0f - r) / TAU_DEPR - BETA * u * r * ht) * DT_C;
            float u_new = u + ((pr - u) / TAU_FACIL + BETA * pr * (1.0f - u) * ht) * DT_C;
            float h_new = h + ((-h + acc[q] + It + offv) / TAU) * DT_C;
            float ht_new = sigm(ACT_GAIN * (h_new - ACT_THRESH));
            rs[q] = r_new; us[q] = u_new; hs[q] = h_new; hts[q] = ht_new;
            preA[m * HP + hidx] = (_Float16)(hv ? (ht_new * r_new * u_new * scl) : 0.0f);
            zpart[q] = ht_new * whz;
            if (hv) {
                size_t o = ((size_t)b * T_STEPS + t) * H_DIM + hidx;
                out[o]                                       = h_new;
                out[(size_t)B_TOT * T_STEPS * H_DIM + o]     = r_new;
                out[2ull * B_TOT * T_STEPS * H_DIM + o]      = u_new;
            }
        }
        // ---- readout z[b] = sum_h ht_new*W_hz: 16-lane xor reduce + LDS atomic ----
#pragma unroll
        for (int q = 0; q < 8; ++q) {
            float v = zpart[q];
            v += __shfl_xor(v, 1, 16);
            v += __shfl_xor(v, 2, 16);
            v += __shfl_xor(v, 4, 16);
            v += __shfl_xor(v, 8, 16);
            zpart[q] = v;
        }
        if (col == 0) {
#pragma unroll
            for (int q = 0; q < 8; ++q)
                atomicAdd(&zbuf[mbase + q], zpart[q]);
        }
        __syncthreads();   // new preA + zbuf complete before next matmul

        if (wv == 0 && tid < MT) {
            int b = bg0 + tid;
            out[3ull * B_TOT * T_STEPS * H_DIM + (size_t)b * T_STEPS + t] = zbuf[tid];
            zbuf[tid] = 0.0f;   // reordered vs. next step's atomics by its first barrier
        }
    }
}

extern "C" void kernel_launch(void* const* d_in, const int* in_sizes, int n_in,
                              void* d_out, int out_size, void* d_ws, size_t ws_size,
                              hipStream_t stream) {
    (void)in_sizes; (void)n_in; (void)d_ws; (void)ws_size; (void)out_size;
    const float* I         = (const float*)d_in[0];
    const float* h0        = (const float*)d_in[1];
    const float* r0        = (const float*)d_in[2];
    const float* u0        = (const float*)d_in[3];
    const float* p_rel     = (const float*)d_in[4];
    const float* offset_ih = (const float*)d_in[5];
    const float* W_hh      = (const float*)d_in[6];
    const float* W_hz      = (const float*)d_in[7];
    float* out = (float*)d_out;

    constexpr size_t lds_bytes = (size_t)(KC * NT * 512 + MT * HP) * sizeof(_Float16)
                               + MT * sizeof(float);   // ~205 KB of the 320 KB WGP LDS
    dim3 grid(B_TOT / MT);          // 2 workgroups, one per 16-row batch tile
    dim3 block(NTHREADS);           // 19 waves, one WMMA N-tile each
    tm_rnn_persistent<<<grid, block, lds_bytes, stream>>>(
        I, h0, r0, u0, p_rel, offset_ih, W_hh, W_hz, out);
}
